// TokenSelection_21079699488982
// MI455X (gfx1250) — matrix operand order repaired and checked by reading, not compiled, not run on hardware
//
#include <hip/hip_runtime.h>

// Problem constants from the reference
#define B_ 8
#define T_ 4096
#define H_ 1024
#define K_ 1024   // T * 0.25
#define MT_ 64    // token rows per block (4 row-tiles of 16)

typedef __attribute__((ext_vector_type(16))) __bf16 v16bf;
typedef __attribute__((ext_vector_type(8)))  float  v8f;

__device__ __forceinline__ unsigned short f2bf(float f) {
    // round-to-nearest-even f32 -> bf16
    unsigned u = __float_as_uint(f);
    u += 0x7FFFu + ((u >> 16) & 1u);
    return (unsigned short)(u >> 16);
}

union FragBF {
    v16bf v;
    unsigned short u[16];
    uint4 q[2];
};

// ---------------------------------------------------------------------------
// Kernel 0: one-shot W1 f32 -> bf16 (2 MB, stays L2-resident for kernel 1)
// ---------------------------------------------------------------------------
__global__ __launch_bounds__(256) void w1_to_bf16(
    const float* __restrict__ W1, unsigned short* __restrict__ W1b) {
    int i = blockIdx.x * 256 + threadIdx.x;     // 4 elements per thread
    float4 f = ((const float4*)W1)[i];
    uint2 o;
    o.x = (unsigned)f2bf(f.x) | ((unsigned)f2bf(f.y) << 16);
    o.y = (unsigned)f2bf(f.z) | ((unsigned)f2bf(f.w) << 16);
    ((uint2*)W1b)[i] = o;
}

// ---------------------------------------------------------------------------
// Kernel 1: scores[b,t] = relu(x[b,t,:] @ W1^T + b1) . W2 + b2
// 256-thread block (8 wave32) per 64-token tile. Each wave owns 8 o-tiles;
// per o-tile it runs 4 WMMAs per k-chunk (4 row-tiles share one B fragment).
// ---------------------------------------------------------------------------
__global__ __launch_bounds__(256) void score_kernel(
    const float* __restrict__ x, const unsigned short* __restrict__ W1b,
    const float* __restrict__ b1, const float* __restrict__ W2,
    const float* __restrict__ b2, float* __restrict__ scores) {

    __shared__ __align__(16) unsigned short lds_x[MT_][H_ + 8]; // pad rows
    __shared__ float lds_s[MT_];

    const int tid = threadIdx.x;
    const int b   = blockIdx.x >> 6;           // T_/MT_ = 64 tiles per batch
    const int t0  = (blockIdx.x & 63) << 6;    // first token of this tile

    // Stage 64x1024 fp32 x-tile into LDS as bf16 (coalesced b128 loads)
    const float4* xsrc = (const float4*)(x + ((size_t)b * T_ + t0) * H_);
    for (int i = tid; i < MT_ * (H_ / 4); i += 256) {
        int row = i >> 8;            // H_/4 = 256 float4 per row
        int c4  = i & 255;
        float4 f = xsrc[(size_t)row * (H_ / 4) + c4];
        int kp = c4 * 4;
        lds_x[row][kp + 0] = f2bf(f.x);
        lds_x[row][kp + 1] = f2bf(f.y);
        lds_x[row][kp + 2] = f2bf(f.z);
        lds_x[row][kp + 3] = f2bf(f.w);
    }
    if (tid < MT_) lds_s[tid] = 0.f;
    __syncthreads();

    const int wave  = tid >> 5;
    const int lane  = tid & 31;
    const int nrow  = lane & 15;   // A: token row in tile / B: output column
    const int khalf = lane >> 4;

    float part[4][8];
    #pragma unroll
    for (int rt = 0; rt < 4; ++rt)
        #pragma unroll
        for (int r = 0; r < 8; ++r) part[rt][r] = 0.f;

    for (int i = 0; i < 8; ++i) {
        const int ot = wave * 8 + i;            // 64 o-tiles / 8 waves
        const int o  = ot * 16 + nrow;          // output channel for this lane
        const unsigned short* w16 = W1b + (size_t)o * H_;
        const float b1v = b1[o];
        const float w2v = W2[o];

        v8f acc0 = {}, acc1 = {}, acc2 = {}, acc3 = {};
        for (int kc = 0; kc < H_; kc += 32) {
            // B fragment (bf16, contiguous k order: per-lane K = 16*khalf + e)
            FragBF bm;
            bm.q[0] = *(const uint4*)(w16 + kc + khalf * 16);
            bm.q[1] = *(const uint4*)(w16 + kc + khalf * 16 + 8);
            __builtin_prefetch((const void*)(w16 + kc + 128), 0, 1);

            // A fragments from LDS (per-lane K = 16*khalf + e, matches B)
            const int kA0 = kc + khalf * 16;
            const int kA1 = kc + khalf * 16 + 8;
            FragBF a0, a1, a2, a3;
            a0.q[0] = *(const uint4*)&lds_x[nrow     ][kA0];
            a0.q[1] = *(const uint4*)&lds_x[nrow     ][kA1];
            a1.q[0] = *(const uint4*)&lds_x[16 + nrow][kA0];
            a1.q[1] = *(const uint4*)&lds_x[16 + nrow][kA1];
            a2.q[0] = *(const uint4*)&lds_x[32 + nrow][kA0];
            a2.q[1] = *(const uint4*)&lds_x[32 + nrow][kA1];
            a3.q[0] = *(const uint4*)&lds_x[48 + nrow][kA0];
            a3.q[1] = *(const uint4*)&lds_x[48 + nrow][kA1];

            acc0 = __builtin_amdgcn_wmma_f32_16x16x32_bf16(
                false, a0.v, false, bm.v, (short)0, acc0, false, false);
            acc1 = __builtin_amdgcn_wmma_f32_16x16x32_bf16(
                false, a1.v, false, bm.v, (short)0, acc1, false, false);
            acc2 = __builtin_amdgcn_wmma_f32_16x16x32_bf16(
                false, a2.v, false, bm.v, (short)0, acc2, false, false);
            acc3 = __builtin_amdgcn_wmma_f32_16x16x32_bf16(
                false, a3.v, false, bm.v, (short)0, acc3, false, false);
        }

        // Fold this o-tile's h values into per-lane score partials (no
        // shuffles here — cross-lane reduction happens once, after the loop).
        #pragma unroll
        for (int r = 0; r < 8; ++r) {
            part[0][r] += fmaxf(acc0[r] + b1v, 0.f) * w2v;
            part[1][r] += fmaxf(acc1[r] + b1v, 0.f) * w2v;
            part[2][r] += fmaxf(acc2[r] + b1v, 0.f) * w2v;
            part[3][r] += fmaxf(acc3[r] + b1v, 0.f) * w2v;
        }
    }

    // One cross-lane reduction per wave (sum over the 16 columns in lanes),
    // then a single LDS atomic per (row-tile, row).
    #pragma unroll
    for (int rt = 0; rt < 4; ++rt) {
        #pragma unroll
        for (int r = 0; r < 8; ++r) {
            float v = part[rt][r];
            v += __shfl_xor(v, 1, 32);
            v += __shfl_xor(v, 2, 32);
            v += __shfl_xor(v, 4, 32);
            v += __shfl_xor(v, 8, 32);
            if (nrow == 0) atomicAdd(&lds_s[rt * 16 + khalf * 8 + r], v);
        }
    }
    __syncthreads();
    if (tid < MT_)
        scores[(size_t)b * T_ + t0 + tid] = lds_s[tid] + b2[0];
}

// ---------------------------------------------------------------------------
// Kernel 2: per-batch full bitonic sort of (score,index) pairs in LDS.
// Comparator: score desc, index asc -> matches top_k sorted/tie semantics.
// ---------------------------------------------------------------------------
__global__ __launch_bounds__(1024) void topk_kernel(
    const float* __restrict__ scores, int* __restrict__ idx_out) {

    __shared__ float key[T_];
    __shared__ int   kid[T_];
    const int tid = threadIdx.x;
    const int b   = blockIdx.x;

    for (int i = tid; i < T_; i += 1024) {
        key[i] = scores[(size_t)b * T_ + i];
        kid[i] = i;
    }
    __syncthreads();

    for (int size = 2; size <= T_; size <<= 1) {
        for (int stride = size >> 1; stride > 0; stride >>= 1) {
            for (int i = tid; i < T_; i += 1024) {
                int j = i ^ stride;
                if (j > i) {
                    float ka = key[i], kb = key[j];
                    int   ia = kid[i], ib = kid[j];
                    bool aFirst = (ka > kb) || (ka == kb && ia < ib);
                    bool desc   = ((i & size) == 0);
                    if (desc ? !aFirst : aFirst) {
                        key[i] = kb; key[j] = ka;
                        kid[i] = ib; kid[j] = ia;
                    }
                }
            }
            __syncthreads();
        }
    }
    for (int j = tid; j < K_; j += 1024)
        idx_out[(size_t)b * K_ + j] = kid[j];
}

// ---------------------------------------------------------------------------
// Kernel 3: gather selected fp32 rows (pure b128 bandwidth).
// ---------------------------------------------------------------------------
__global__ __launch_bounds__(256) void gather_kernel(
    const float* __restrict__ x, const int* __restrict__ idx,
    float* __restrict__ out) {

    const int b = blockIdx.x / K_;
    const int j = blockIdx.x % K_;
    int row = idx[(size_t)b * K_ + j] & (T_ - 1);
    const float4* src = (const float4*)(x + ((size_t)b * T_ + row) * H_);
    float4*       dst = (float4*)(out + ((size_t)b * K_ + j) * H_);
    dst[threadIdx.x] = src[threadIdx.x];   // 256 threads * float4 = 1024 floats
}

// ---------------------------------------------------------------------------
extern "C" void kernel_launch(void* const* d_in, const int* in_sizes, int n_in,
                              void* d_out, int out_size, void* d_ws, size_t ws_size,
                              hipStream_t stream) {
    const float* x  = (const float*)d_in[0];
    // d_in[1] = attention_scores (unused by the reference)
    const float* W1 = (const float*)d_in[2];
    const float* b1 = (const float*)d_in[3];
    const float* W2 = (const float*)d_in[4];
    const float* b2 = (const float*)d_in[5];

    float* tokens  = (float*)d_out;                          // [B, K, H] fp32
    int*   idx_out = (int*)(tokens + (size_t)B_ * K_ * H_);  // [B, K] int32 tail
    float* scores  = (float*)d_ws;                           // [B, T] fp32 scratch
    unsigned short* W1b = (unsigned short*)((char*)d_ws + (size_t)B_ * T_ * 4);

    w1_to_bf16   <<< (H_ * H_ / 4) / 256, 256, 0, stream >>> (W1, W1b);
    score_kernel <<< B_ * (T_ / MT_),     256, 0, stream >>> (x, W1b, b1, W2, b2, scores);
    topk_kernel  <<< B_,                 1024, 0, stream >>> (scores, idx_out);
    gather_kernel<<< B_ * K_,             256, 0, stream >>> (x, idx_out, tokens);

    (void)in_sizes; (void)n_in; (void)out_size; (void)ws_size;
}
